// PointNetSetAbstractionMsg_46557445488934
// MI455X (gfx1250) — compile-verified, matrix operand order checked by reading, not run on hardware
//
#include <hip/hip_runtime.h>
#include <hip/hip_bf16.h>

// ---------------------------------------------------------------------------
// PointNet++ Set Abstraction (MSG) for MI455X / gfx1250
//   B=8, N=4096, S=1024, Cin=64, radii (.1,.2,.4), K=(16,32,128)
//   MLPs: (32,32,64) (64,64,128) (64,96,128), BN(train) + ReLU, maxpool K
// GEMMs run on v_wmma_f32_16x16x32_f16 (f16 in, f32 accumulate).
// ---------------------------------------------------------------------------

typedef __attribute__((ext_vector_type(16))) _Float16 v16h;
typedef __attribute__((ext_vector_type(8)))  float    v8f;

#define BB   8
#define NPTS 4096
#define SS   1024
#define CIN  64
#define OUT_XYZ_FLOATS (BB * SS * 3)      // 24576
#define OUT_FEAT_CH    320

// ===========================================================================
// 1) Farthest point sampling: one block per batch, 256 threads, 16 pts/thread
//    Emits new_xyz [B,S,3] directly into d_out (first 24576 floats).
// ===========================================================================
__global__ void fps_kernel(const float* __restrict__ xyz, float* __restrict__ newxyz)
{
    __shared__ float rv[256];
    __shared__ int   ri[256];
    __shared__ float cent[3];

    const int b = blockIdx.x, tid = threadIdx.x;
    const float* xb = xyz + (size_t)b * NPTS * 3;

    float px[16], py[16], pz[16], dist[16];
#pragma unroll
    for (int j = 0; j < 16; ++j) {
        int p = tid + j * 256;
        px[j] = xb[p * 3 + 0];
        py[j] = xb[p * 3 + 1];
        pz[j] = xb[p * 3 + 2];
        dist[j] = 1e10f;
    }

    int far = 0;  // scan starts with farthest = 0
    for (int s = 0; s < SS; ++s) {
        if (tid == (far & 255)) {
            int j = far >> 8;
            cent[0] = px[j]; cent[1] = py[j]; cent[2] = pz[j];
        }
        __syncthreads();
        float cx = cent[0], cy = cent[1], cz = cent[2];
        if (tid == 0) {
            size_t o = ((size_t)b * SS + s) * 3;
            newxyz[o + 0] = cx; newxyz[o + 1] = cy; newxyz[o + 2] = cz;
        }

        float bv = -1.0f; int bi = 0;
#pragma unroll
        for (int j = 0; j < 16; ++j) {
            float dx = px[j] - cx, dy = py[j] - cy, dz = pz[j] - cz;
            float d  = dx * dx + dy * dy + dz * dz;
            float nd = fminf(dist[j], d);
            dist[j] = nd;
            if (nd > bv) { bv = nd; bi = tid + j * 256; }  // ascending -> first max
        }
        rv[tid] = bv; ri[tid] = bi;
        __syncthreads();
        for (int off = 128; off > 0; off >>= 1) {
            if (tid < off) {
                float v2 = rv[tid + off]; int i2 = ri[tid + off];
                if (v2 > rv[tid] || (v2 == rv[tid] && i2 < ri[tid])) {
                    rv[tid] = v2; ri[tid] = i2;
                }
            }
            __syncthreads();
        }
        far = ri[0];
        __syncthreads();   // protect rv/ri/cent reuse next iteration
    }
}

// ===========================================================================
// 2) Ball query (all 3 branches): one block per (b,s), 128 threads.
//    d2 computed once in LDS; per branch mask + K-pass min-extract with
//    lower-index tie break; out-of-radius slots fall back to first neighbor.
// ===========================================================================
__global__ void ballquery_kernel(const float* __restrict__ xyz,
                                 const float* __restrict__ newxyz,
                                 int* __restrict__ idx0,
                                 int* __restrict__ idx1,
                                 int* __restrict__ idx2)
{
    __shared__ float d2[NPTS];
    __shared__ float wk[NPTS];
    __shared__ float rv[128];
    __shared__ int   ri[128];
    __shared__ int   seli[128];
    __shared__ float selv[128];

    const int tid = threadIdx.x;
    const int bs  = blockIdx.x;           // b*S + s
    const int b   = bs >> 10;

    const float cx = newxyz[bs * 3 + 0];
    const float cy = newxyz[bs * 3 + 1];
    const float cz = newxyz[bs * 3 + 2];
    const float* xb = xyz + (size_t)b * NPTS * 3;

    for (int i = tid; i < NPTS; i += 128) {
        float dx = xb[i * 3 + 0] - cx;
        float dy = xb[i * 3 + 1] - cy;
        float dz = xb[i * 3 + 2] - cz;
        d2[i] = dx * dx + dy * dy + dz * dz;
    }
    __syncthreads();

    const float r2s[3] = {0.01f, 0.04f, 0.16f};
    const int   ks[3]  = {16, 32, 128};
    int* outs[3] = {idx0, idx1, idx2};

    for (int br = 0; br < 3; ++br) {
        const float r2 = r2s[br];
        const int   Kn = ks[br];
        for (int i = tid; i < NPTS; i += 128)
            wk[i] = (d2[i] <= r2) ? d2[i] : __builtin_inff();
        __syncthreads();

        for (int j = 0; j < Kn; ++j) {
            float bv = __builtin_inff(); int bi = 0x7fffffff;
            for (int i = tid; i < NPTS; i += 128) {
                float v = wk[i];
                if (v < bv) { bv = v; bi = i; }   // ascending keeps lowest idx
            }
            rv[tid] = bv; ri[tid] = bi;
            __syncthreads();
            for (int off = 64; off > 0; off >>= 1) {
                if (tid < off) {
                    float v2 = rv[tid + off]; int i2 = ri[tid + off];
                    if (v2 < rv[tid] || (v2 == rv[tid] && i2 < ri[tid])) {
                        rv[tid] = v2; ri[tid] = i2;
                    }
                }
                __syncthreads();
            }
            if (tid == 0) {
                seli[j] = ri[0]; selv[j] = rv[0];
                if (ri[0] < NPTS) wk[ri[0]] = __builtin_inff();
            }
            __syncthreads();
        }

        int* ob = outs[br] + (size_t)bs * Kn;
        for (int j = tid; j < Kn; j += 128)
            ob[j] = (selv[j] <= r2) ? seli[j] : seli[0];
        __syncthreads();
    }
}

// ===========================================================================
// 3) Gather -> f16 X0 [M, 96]: c<3 rel-xyz, 3..66 features, 67..95 zero pad
// ===========================================================================
__global__ void gather_kernel(const float* __restrict__ xyz,
                              const float* __restrict__ feat,
                              const float* __restrict__ newxyz,
                              const int* __restrict__ idx,
                              _Float16* __restrict__ X0,
                              int Kn, long Mrows)
{
    long e = (long)blockIdx.x * blockDim.x + threadIdx.x;
    if (e >= Mrows * 96) return;
    long m = e / 96;
    int  c = (int)(e - m * 96);
    long bs = m / Kn;                  // b*S + s
    int  b  = (int)(bs >> 10);
    int  id = idx[m];

    float v;
    if (c < 3)
        v = xyz[((size_t)b * NPTS + id) * 3 + c] - newxyz[bs * 3 + c];
    else if (c < 67)
        v = feat[((size_t)b * NPTS + id) * CIN + (c - 3)];
    else
        v = 0.0f;
    X0[e] = (_Float16)v;
}

// ===========================================================================
// 4) Weight f32 -> padded f16 [oc, Kp]
// ===========================================================================
__global__ void convw_kernel(const float* __restrict__ W, _Float16* __restrict__ dst,
                             int oc, int last, int Kp)
{
    int e = blockIdx.x * blockDim.x + threadIdx.x;
    if (e >= oc * Kp) return;
    int o = e / Kp, k = e - o * Kp;
    dst[e] = (_Float16)((k < last) ? W[o * last + k] : 0.0f);
}

// ===========================================================================
// 5) GEMM  Y[M,N] = X[M,K] * Wt[N,K]^T + bias, f16 in / f32 acc via WMMA,
//    plus per-channel sum / sumsq for BatchNorm batch statistics.
//    One wave per 16-row tile; NT = N/16 accumulators held per wave.
//    A/B lane layout per CDNA5 16-bit 16x32 A-matrix table:
//      lane l: row = l&15, halves = K[off..off+8) ++ K[off+16..off+24),
//      off = (l<16) ? 0 : 8  -> two 16B global_load_b128 per operand.
// ===========================================================================
union ABu { v16h v; uint4 u[2]; };

template <int NT>
__global__ void gemm_bias_stats(const _Float16* __restrict__ X,
                                const _Float16* __restrict__ Wt,
                                const float* __restrict__ bias,
                                _Float16* __restrict__ Y,
                                float* __restrict__ stats,
                                int K)
{
    constexpr int N = NT * 16;
    __shared__ float s_sum[N];
    __shared__ float s_sq[N];

    const int tid = threadIdx.x;
    for (int i = tid; i < N; i += 256) { s_sum[i] = 0.0f; s_sq[i] = 0.0f; }
    __syncthreads();

    const int lane = tid & 31;
    const int wave = tid >> 5;
    const long row0 = ((long)blockIdx.x * 8 + wave) * 16;
    const int r    = lane & 15;
    const int hi   = lane >> 4;
    const int koff = hi ? 8 : 0;

    const _Float16* xrow = X + (size_t)(row0 + r) * K;

    v8f acc[NT];
#pragma unroll
    for (int nt = 0; nt < NT; ++nt) acc[nt] = (v8f){};

    for (int k0 = 0; k0 < K; k0 += 32) {
        ABu a;
        a.u[0] = *reinterpret_cast<const uint4*>(xrow + k0 + koff);
        a.u[1] = *reinterpret_cast<const uint4*>(xrow + k0 + koff + 16);
#pragma unroll
        for (int nt = 0; nt < NT; ++nt) {
            const _Float16* wrow = Wt + (size_t)(nt * 16 + r) * K;
            ABu bm;
            bm.u[0] = *reinterpret_cast<const uint4*>(wrow + k0 + koff);
            bm.u[1] = *reinterpret_cast<const uint4*>(wrow + k0 + koff + 16);
            acc[nt] = __builtin_amdgcn_wmma_f32_16x16x32_f16(
                false, a.v, false, bm.v, (short)0, acc[nt], false, false);
        }
    }

    // Epilogue: bias, store f16, per-channel partial stats.
    // C/D layout: vgpr v, lane l -> M = row0 + v + 8*(l>=16), Ncol = nt*16 + (l&15)
#pragma unroll
    for (int nt = 0; nt < NT; ++nt) {
        const int col = nt * 16 + r;
        const float bb = bias[col];
        float lsum = 0.0f, lsq = 0.0f;
#pragma unroll
        for (int v = 0; v < 8; ++v) {
            float y = acc[nt][v] + bb;
            size_t m = (size_t)(row0 + v + 8 * hi);
            Y[m * N + col] = (_Float16)y;
            lsum += y;
            lsq  += y * y;
        }
        atomicAdd(&s_sum[col], lsum);
        atomicAdd(&s_sq[col],  lsq);
    }
    __syncthreads();
    for (int i = tid; i < N; i += 256) {
        atomicAdd(&stats[i],     s_sum[i]);
        atomicAdd(&stats[N + i], s_sq[i]);
    }
}

// ===========================================================================
// 6) BN(train stats) + ReLU, in place, f16
// ===========================================================================
__global__ void bn_relu_inplace(_Float16* __restrict__ Y,
                                const float* __restrict__ stats,
                                const float* __restrict__ gamma,
                                const float* __restrict__ beta,
                                long Mrows, int N)
{
    long e = (long)blockIdx.x * blockDim.x + threadIdx.x;
    if (e >= Mrows * N) return;
    int n = (int)(e % N);
    float Mf   = (float)Mrows;
    float mean = stats[n] / Mf;
    float var  = stats[N + n] / Mf - mean * mean;
    float sc   = gamma[n] * rsqrtf(var + 1e-5f);
    float v    = ((float)Y[e] - mean) * sc + beta[n];
    Y[e] = (_Float16)(v > 0.0f ? v : 0.0f);
}

// ===========================================================================
// 7) Final layer: BN + ReLU + max over K neighbors -> output slice (f32)
// ===========================================================================
__global__ void bn_relu_maxpool(const _Float16* __restrict__ Y,
                                const float* __restrict__ stats,
                                const float* __restrict__ gamma,
                                const float* __restrict__ beta,
                                float* __restrict__ out,
                                int Kn, int N, int coloff)
{
    const int BSQ = BB * SS;
    int e = blockIdx.x * blockDim.x + threadIdx.x;
    if (e >= BSQ * N) return;
    int bs = e / N, n = e - bs * N;

    float Mf   = (float)BSQ * (float)Kn;
    float mean = stats[n] / Mf;
    float var  = stats[N + n] / Mf - mean * mean;
    float sc   = gamma[n] * rsqrtf(var + 1e-5f);
    float bt   = beta[n];

    const _Float16* p = Y + (size_t)bs * Kn * N + n;
    float mx = -3.4e38f;
    for (int k = 0; k < Kn; ++k) {
        float v = ((float)p[(size_t)k * N] - mean) * sc + bt;
        v = v > 0.0f ? v : 0.0f;
        mx = fmaxf(mx, v);
    }
    out[OUT_XYZ_FLOATS + (size_t)bs * OUT_FEAT_CH + coloff + n] = mx;
}

// ===========================================================================
// Host orchestration
// ===========================================================================
extern "C" void kernel_launch(void* const* d_in, const int* in_sizes, int n_in,
                              void* d_out, int out_size, void* d_ws, size_t ws_size,
                              hipStream_t stream)
{
    (void)in_sizes; (void)n_in; (void)out_size; (void)ws_size;

    const float* xyz  = (const float*)d_in[0];
    const float* feat = (const float*)d_in[1];
    float* out = (float*)d_out;

    static const int mlps[3][3]  = {{32, 32, 64}, {64, 64, 128}, {64, 96, 128}};
    static const int nsamp[3]    = {16, 32, 128};
    static const int coloff[3]   = {0, 64, 192};

    // ---- workspace carve-out (256B aligned) ----
    char* ws = (char*)d_ws;
    size_t off = 0;
    auto take = [&](size_t bytes) -> char* {
        char* p = ws + off;
        off = (off + bytes + 255) & ~(size_t)255;
        return p;
    };
    _Float16* wf16  = (_Float16*)take((size_t)9 * 128 * 128 * sizeof(_Float16));
    float*    stats = (float*)take(2 * 128 * sizeof(float));
    int* idxb[3];
    idxb[0] = (int*)take((size_t)BB * SS * 16  * sizeof(int));
    idxb[1] = (int*)take((size_t)BB * SS * 32  * sizeof(int));
    idxb[2] = (int*)take((size_t)BB * SS * 128 * sizeof(int));
    _Float16* buf0 = (_Float16*)take((size_t)BB * SS * 128 * 96  * sizeof(_Float16));
    _Float16* buf1 = (_Float16*)take((size_t)BB * SS * 128 * 128 * sizeof(_Float16));

    // ---- convert weights to padded f16 ----
    for (int br = 0; br < 3; ++br)
        for (int l = 0; l < 3; ++l) {
            const float* W = (const float*)d_in[2 + (br * 3 + l) * 4 + 0];
            int oc   = mlps[br][l];
            int last = (l == 0) ? 67 : mlps[br][l - 1];
            int Kp   = (l == 0) ? 96 : last;
            _Float16* dst = wf16 + (size_t)(br * 3 + l) * 128 * 128;
            int tot = oc * Kp;
            convw_kernel<<<(tot + 255) / 256, 256, 0, stream>>>(W, dst, oc, last, Kp);
        }

    // ---- FPS -> new_xyz directly into d_out ----
    fps_kernel<<<BB, 256, 0, stream>>>(xyz, out);

    // ---- ball query for all 3 branches ----
    ballquery_kernel<<<BB * SS, 128, 0, stream>>>(xyz, out, idxb[0], idxb[1], idxb[2]);

    // ---- per-branch gather + MLP (WMMA GEMM + BN-ReLU) + maxpool ----
    for (int br = 0; br < 3; ++br) {
        const int  Kn    = nsamp[br];
        const long Mrows = (long)BB * SS * Kn;   // 131072 / 262144 / 1048576

        long tot0 = Mrows * 96;
        gather_kernel<<<(unsigned)((tot0 + 255) / 256), 256, 0, stream>>>(
            xyz, feat, out, idxb[br], buf0, Kn, Mrows);

        _Float16* bufs[2] = {buf0, buf1};
        int cur = 0;
        for (int l = 0; l < 3; ++l) {
            const int Kp = (l == 0) ? 96 : mlps[br][l - 1];
            const int Nc = mlps[br][l];
            const float* bias  = (const float*)d_in[2 + (br * 3 + l) * 4 + 1];
            const float* gamma = (const float*)d_in[2 + (br * 3 + l) * 4 + 2];
            const float* beta  = (const float*)d_in[2 + (br * 3 + l) * 4 + 3];
            const _Float16* Wt = wf16 + (size_t)(br * 3 + l) * 128 * 128;

            hipMemsetAsync(stats, 0, 2 * Nc * sizeof(float), stream);

            dim3 g((unsigned)(Mrows / 128)), blk(256);  // 8 waves/block, 1 mtile/wave
            switch (Nc / 16) {
            case 2: gemm_bias_stats<2><<<g, blk, 0, stream>>>(bufs[cur], Wt, bias, bufs[1 - cur], stats, Kp); break;
            case 4: gemm_bias_stats<4><<<g, blk, 0, stream>>>(bufs[cur], Wt, bias, bufs[1 - cur], stats, Kp); break;
            case 6: gemm_bias_stats<6><<<g, blk, 0, stream>>>(bufs[cur], Wt, bias, bufs[1 - cur], stats, Kp); break;
            case 8: gemm_bias_stats<8><<<g, blk, 0, stream>>>(bufs[cur], Wt, bias, bufs[1 - cur], stats, Kp); break;
            }
            cur ^= 1;

            if (l < 2) {
                long tot = Mrows * Nc;
                bn_relu_inplace<<<(unsigned)((tot + 255) / 256), 256, 0, stream>>>(
                    bufs[cur], stats, gamma, beta, Mrows, Nc);
            } else {
                int tot = BB * SS * Nc;
                bn_relu_maxpool<<<(tot + 255) / 256, 256, 0, stream>>>(
                    bufs[cur], stats, gamma, beta, out, Kn, Nc, coloff[br]);
            }
        }
    }
}
